// GAT_24979529793768
// MI455X (gfx1250) — matrix operand order repaired and checked by reading, not compiled, not run on hardware
//
#include <hip/hip_runtime.h>
#include <hip/hip_bf16.h>
#include <math.h>

typedef _Float16 half_t;
typedef __attribute__((ext_vector_type(16))) _Float16 v16h;
typedef __attribute__((ext_vector_type(8)))  _Float16 v8h;
typedef __attribute__((ext_vector_type(4)))  _Float16 v4h;
typedef __attribute__((ext_vector_type(8)))  float    v8f;
typedef __attribute__((ext_vector_type(4)))  float    v4f;

#define BB_ 8
#define NN_ 512
#define DD_ 1024
#define TT_ (BB_*NN_)   // 4096 tokens

__device__ __forceinline__ v8f wmma_f16(v16h a, v16h b, v8f c) {
  // v_wmma_f32_16x16x32_f16: D = A(16x32 f16) x B(32x16 f16) + C(16x16 f32)
  return __builtin_amdgcn_wmma_f32_16x16x32_f16(
      /*neg_a=*/false, a, /*neg_b=*/false, b,
      /*c_mod=*/(short)0, c, /*reuse_a=*/false, /*reuse_b=*/false);
}

// ---- Stage 1a: h (f32, row-major [4096][1024]) -> f16 row-major ----
__global__ void cvt_h(const float* __restrict__ h, half_t* __restrict__ out) {
  int i = (blockIdx.x * blockDim.x + threadIdx.x) * 4;
  v4f x = *(const v4f*)(h + i);
  v4h o;
  o[0] = (half_t)x[0]; o[1] = (half_t)x[1];
  o[2] = (half_t)x[2]; o[3] = (half_t)x[3];
  *(v4h*)(out + i) = o;
}

// ---- Stage 1b: W (f32, [k][n]) -> f16 transposed Wt[n][k], LDS-tiled so both
// the global read and the global write are coalesced (avoids 128B/4B reads). ----
__global__ void __launch_bounds__(256) cvt_wt(const float* __restrict__ W,
                                              half_t* __restrict__ Wt) {
  __shared__ half_t tile[32][33];               // +1 pad: no bank conflicts
  const int n0 = blockIdx.x * 32;
  const int k0 = blockIdx.y * 32;
  const int tx = threadIdx.x, ty = threadIdx.y; // 32 x 8
#pragma unroll
  for (int i = 0; i < 4; ++i) {                 // read W[k][n], n contiguous
    int k = k0 + ty + i * 8;
    tile[tx][ty + i * 8] = (half_t)W[(size_t)k * DD_ + n0 + tx];
  }
  __syncthreads();
#pragma unroll
  for (int i = 0; i < 4; ++i) {                 // write Wt[n][k], k contiguous
    int n = n0 + ty + i * 8;
    Wt[(size_t)n * DD_ + k0 + tx] = tile[ty + i * 8][tx];
  }
}

// ---- Stage 2: s[b,i,j] from adj / adj_weight (the collapsed softmax) ----
__global__ void build_s(const int* __restrict__ adj, const float* __restrict__ adjw,
                        half_t* __restrict__ s) {
  int row = blockIdx.x;                 // 0..4095 = b*512 + i
  const int*   arow = adj  + (size_t)row * NN_;
  const float* wrow = adjw + (size_t)row * NN_;
  half_t*      srow = s    + (size_t)row * NN_;
  int tid = threadIdx.x;                // 256 threads, 2 cols each
  int a0 = (arow[tid]       != 0);
  int a1 = (arow[tid + 256] != 0);
  __shared__ int red[256];
  red[tid] = a0 + a1;
  __syncthreads();
  for (int off = 128; off > 0; off >>= 1) {
    if (tid < off) red[tid] += red[tid + off];
    __syncthreads();
  }
  int cnt = red[0];
  float inv = (cnt > 0) ? (1.0f / (float)cnt) : (1.0f / (float)NN_);
  float v0 = (cnt == 0 || a0) ? wrow[tid]       * inv : 0.0f;
  float v1 = (cnt == 0 || a1) ? wrow[tid + 256] * inv : 0.0f;
  srow[tid]       = (half_t)v0;
  srow[tid + 256] = (half_t)v1;
}

// ---- Stage 3: GEMM1  Wh = h @ W, output written transposed as WhT[b][d][n] (f16)
// Wave computes 16(M=token) x 64(N=d); K=1024 in steps of 32, ping-pong
// double-buffered so global loads overlap the WMMA bursts. ----
__global__ void __launch_bounds__(128) gemm1(const half_t* __restrict__ A,   // [4096][1024]
                                             const half_t* __restrict__ Bt,  // [1024][1024] = W^T
                                             half_t* __restrict__ WhT) {     // [8][1024][512]
  const int lane  = threadIdx.x & 31;
  const int wave  = threadIdx.x >> 5;
  const int mtile = blockIdx.x;                 // 0..255
  const int ntile = blockIdx.y * 4 + wave;      // 0..15 (64 cols each)
  const int mrow  = lane & 15;
  const int hi    = lane >> 4;
  const int hi8   = hi * 8;

  const half_t* ap  = A  + (size_t)(mtile * 16 + mrow) * DD_;
  const half_t* bp0 = Bt + (size_t)(ntile * 64 + mrow) * DD_ + hi * 16;
  const half_t* bp1 = bp0 + (size_t)16 * DD_;
  const half_t* bp2 = bp0 + (size_t)32 * DD_;
  const half_t* bp3 = bp0 + (size_t)48 * DD_;

  v8f acc0 = {}, acc1 = {}, acc2 = {}, acc3 = {};

  auto load_frags = [&](int k, v16h& a_, v16h& b0_, v16h& b1_, v16h& b2_, v16h& b3_) {
    ((v8h*)&a_)[0] = *(const v8h*)(ap + k + hi * 16);     // K = k + hi*16 + 0..7
    ((v8h*)&a_)[1] = *(const v8h*)(ap + k + hi * 16 + 8); // K = k + hi*16 + 8..15
    b0_ = *(const v16h*)(bp0 + k);
    b1_ = *(const v16h*)(bp1 + k);
    b2_ = *(const v16h*)(bp2 + k);
    b3_ = *(const v16h*)(bp3 + k);
  };

  v16h aX, b0X, b1X, b2X, b3X;
  v16h aY, b0Y, b1Y, b2Y, b3Y;
  load_frags(0, aX, b0X, b1X, b2X, b3X);
  int k0 = 0;
  for (; k0 + 64 < DD_; k0 += 64) {
    __builtin_prefetch(bp0 + k0 + 128, 0, 1);            // global_prefetch_b8
    __builtin_prefetch(ap  + k0 + 128, 0, 1);
    load_frags(k0 + 32, aY, b0Y, b1Y, b2Y, b3Y);
    acc0 = wmma_f16(aX, b0X, acc0);
    acc1 = wmma_f16(aX, b1X, acc1);
    acc2 = wmma_f16(aX, b2X, acc2);
    acc3 = wmma_f16(aX, b3X, acc3);
    load_frags(k0 + 64, aX, b0X, b1X, b2X, b3X);
    acc0 = wmma_f16(aY, b0Y, acc0);
    acc1 = wmma_f16(aY, b1Y, acc1);
    acc2 = wmma_f16(aY, b2Y, acc2);
    acc3 = wmma_f16(aY, b3Y, acc3);
  }
  load_frags(k0 + 32, aY, b0Y, b1Y, b2Y, b3Y);
  acc0 = wmma_f16(aX, b0X, acc0);
  acc1 = wmma_f16(aX, b1X, acc1);
  acc2 = wmma_f16(aX, b2X, acc2);
  acc3 = wmma_f16(aX, b3X, acc3);
  acc0 = wmma_f16(aY, b0Y, acc0);
  acc1 = wmma_f16(aY, b1Y, acc1);
  acc2 = wmma_f16(aY, b2Y, acc2);
  acc3 = wmma_f16(aY, b3Y, acc3);

  // D frag: lane -> col d, VGPR v -> token t = mtile*16 + 8*hi + v (8 consecutive).
  // Write f16 into WhT[b][d][n] -> contiguous 8-elem store (free transpose for GEMM2).
  int t0 = mtile * 16 + hi * 8;
  int b  = t0 >> 9;
  int n0 = t0 & (NN_ - 1);
  v8f accs[4] = {acc0, acc1, acc2, acc3};
#pragma unroll
  for (int j = 0; j < 4; ++j) {
    int d = ntile * 64 + j * 16 + mrow;
    v8h o;
#pragma unroll
    for (int v = 0; v < 8; ++v) o[v] = (half_t)accs[j][v];
    *(v8h*)(WhT + ((size_t)b * DD_ + d) * NN_ + n0) = o;
  }
}

// ---- Stage 4: GEMM2  out = elu(s @ Wh) per batch; K=512, same pipeline ----
__global__ void __launch_bounds__(128) gemm2(const half_t* __restrict__ S,    // [8][512][512]
                                             const half_t* __restrict__ WhT,  // [8][1024][512]
                                             float* __restrict__ out) {       // [8][512][1024]
  const int lane  = threadIdx.x & 31;
  const int wave  = threadIdx.x >> 5;
  const int b     = blockIdx.z;                 // 0..7
  const int itile = blockIdx.x;                 // 0..31
  const int ntile = blockIdx.y * 4 + wave;      // 0..15
  const int mrow  = lane & 15;
  const int hi    = lane >> 4;
  const int hi8   = hi * 8;

  const half_t* ap  = S   + ((size_t)b * NN_ + itile * 16 + mrow) * NN_;
  const half_t* bp0 = WhT + ((size_t)b * DD_ + ntile * 64 + mrow) * NN_ + hi * 16;
  const half_t* bp1 = bp0 + (size_t)16 * NN_;
  const half_t* bp2 = bp0 + (size_t)32 * NN_;
  const half_t* bp3 = bp0 + (size_t)48 * NN_;

  v8f acc0 = {}, acc1 = {}, acc2 = {}, acc3 = {};

  auto load_frags = [&](int k, v16h& a_, v16h& b0_, v16h& b1_, v16h& b2_, v16h& b3_) {
    ((v8h*)&a_)[0] = *(const v8h*)(ap + k + hi * 16);
    ((v8h*)&a_)[1] = *(const v8h*)(ap + k + hi * 16 + 8);
    b0_ = *(const v16h*)(bp0 + k);
    b1_ = *(const v16h*)(bp1 + k);
    b2_ = *(const v16h*)(bp2 + k);
    b3_ = *(const v16h*)(bp3 + k);
  };

  v16h aX, b0X, b1X, b2X, b3X;
  v16h aY, b0Y, b1Y, b2Y, b3Y;
  load_frags(0, aX, b0X, b1X, b2X, b3X);
  int k0 = 0;
  for (; k0 + 64 < NN_; k0 += 64) {
    __builtin_prefetch(bp0 + k0 + 128, 0, 1);
    __builtin_prefetch(ap  + k0 + 128, 0, 1);
    load_frags(k0 + 32, aY, b0Y, b1Y, b2Y, b3Y);
    acc0 = wmma_f16(aX, b0X, acc0);
    acc1 = wmma_f16(aX, b1X, acc1);
    acc2 = wmma_f16(aX, b2X, acc2);
    acc3 = wmma_f16(aX, b3X, acc3);
    load_frags(k0 + 64, aX, b0X, b1X, b2X, b3X);
    acc0 = wmma_f16(aY, b0Y, acc0);
    acc1 = wmma_f16(aY, b1Y, acc1);
    acc2 = wmma_f16(aY, b2Y, acc2);
    acc3 = wmma_f16(aY, b3Y, acc3);
  }
  load_frags(k0 + 32, aY, b0Y, b1Y, b2Y, b3Y);
  acc0 = wmma_f16(aX, b0X, acc0);
  acc1 = wmma_f16(aX, b1X, acc1);
  acc2 = wmma_f16(aX, b2X, acc2);
  acc3 = wmma_f16(aX, b3X, acc3);
  acc0 = wmma_f16(aY, b0Y, acc0);
  acc1 = wmma_f16(aY, b1Y, acc1);
  acc2 = wmma_f16(aY, b2Y, acc2);
  acc3 = wmma_f16(aY, b3Y, acc3);

  float* op = out + (size_t)b * NN_ * DD_;
  int i0 = itile * 16 + hi * 8;
  v8f accs[4] = {acc0, acc1, acc2, acc3};
#pragma unroll
  for (int j = 0; j < 4; ++j) {
    int d = ntile * 64 + j * 16 + mrow;
#pragma unroll
    for (int v = 0; v < 8; ++v) {
      float x = accs[j][v];
      x = (x > 0.0f) ? x : expm1f(x);           // elu (alpha=1)
      op[(size_t)(i0 + v) * DD_ + d] = x;       // lanes 0..15 -> consecutive d: coalesced
    }
  }
}

extern "C" void kernel_launch(void* const* d_in, const int* in_sizes, int n_in,
                              void* d_out, int out_size, void* d_ws, size_t ws_size,
                              hipStream_t stream) {
  (void)in_sizes; (void)n_in; (void)out_size; (void)ws_size;
  const float* h    = (const float*)d_in[0];   // [8,512,1024]
  const int*   adj  = (const int*)  d_in[1];   // [8,512,512]
  const float* adjw = (const float*)d_in[2];   // [8,512,512]
  const float* W    = (const float*)d_in[3];   // [1,1024,1024]
  // d_in[4] = a : mathematically cancels out of the reference (see analysis)
  float* out = (float*)d_out;                  // [8,512,1024]

  char* w = (char*)d_ws;
  half_t* hA  = (half_t*)(w);                        //  8 MiB: h in f16
  half_t* WtB = (half_t*)(w + ( 8u << 20));          //  2 MiB: W^T in f16
  half_t* WhT = (half_t*)(w + (10u << 20));          //  8 MiB: Wh^T in f16
  half_t* Sf  = (half_t*)(w + (18u << 20));          //  4 MiB: s in f16

  cvt_h  <<<(TT_ * DD_) / (256 * 4), 256, 0, stream>>>(h, hA);
  cvt_wt <<<dim3(DD_ / 32, DD_ / 32), dim3(32, 8), 0, stream>>>(W, WtB);
  gemm1  <<<dim3(TT_ / 16, 4, 1),    128, 0, stream>>>(hA, WtB, WhT);
  build_s<<<TT_,                     256, 0, stream>>>(adj, adjw, Sf);
  gemm2  <<<dim3(NN_ / 16, 4, BB_),  128, 0, stream>>>(Sf, WhT, out);
}
